// GraphAttentionLayer_76965813944553
// MI455X (gfx1250) — compile-verified, matrix operand order checked by reading, not run on hardware
//
#include <hip/hip_runtime.h>
#include <hip/hip_bf16.h>
#include <cstdint>
#include <cstddef>

// ---------------- problem constants (from reference) ----------------
#define N_NODES 100000
#define N_EDGES 1600000
#define IN_CH   128
#define EDGE_CH 16
#define HEADS   4
#define OUT_CH  16
#define HC      64          // HEADS*OUT_CH

// ---------------- workspace layout (bytes) ----------------
#define WS_H       ((size_t)0)                       // h: N*64 f32        = 25,600,000
#define WS_WT      ((size_t)25600000)                // Wt bf16 packed u32 = 16,384
#define WS_AL      ((size_t)25616384)                // alpha_l: N*4 f32   = 1,600,000
#define WS_AR      ((size_t)27216384)                // alpha_r
#define WS_AMAX    ((size_t)28816384)                // segment max
#define WS_DENOM   ((size_t)30416384)                // segment sum of exp
#define WS_ALPHAE  ((size_t)32016384)                // per-edge alpha: E*4 f32 = 25,600,000

typedef __attribute__((ext_vector_type(16))) __bf16 v16bf;
typedef __attribute__((ext_vector_type(8)))  float  v8f;

union ABFrag { v16bf v; unsigned int u[8]; };
union AccFrag { v8f v; float f[8]; };

static __device__ __forceinline__ unsigned short f2bf(float f) {
  unsigned int u = __float_as_uint(f);
  u += 0x7FFFu + ((u >> 16) & 1u);          // round-to-nearest-even
  return (unsigned short)(u >> 16);
}
static __device__ __forceinline__ unsigned int pack_bf2(float lo, float hi) {
  return (unsigned int)f2bf(lo) | ((unsigned int)f2bf(hi) << 16);
}

// float atomic max via signed/unsigned integer ordering trick
static __device__ __forceinline__ void atomicMaxF(float* addr, float val) {
  if (val >= 0.0f) atomicMax((int*)addr, __float_as_int(val));
  else             atomicMin((unsigned int*)addr, __float_as_uint(val));
}

// ---------------- K1: W [128][64] f32 -> Wt [64][64] packed (2 bf16 / u32), K-major ----------------
__global__ void gat_wt_convert(const float* __restrict__ W, unsigned int* __restrict__ Wtu) {
  int i = blockIdx.x * blockDim.x + threadIdx.x;          // i = n*64 + kpair
  if (i >= HC * (IN_CH / 2)) return;
  int n = i >> 6, kp = i & 63, k0 = kp * 2;
  Wtu[i] = pack_bf2(W[(size_t)k0 * HC + n], W[(size_t)(k0 + 1) * HC + n]);
}

// ---------------- K2: h = x @ W via v_wmma_f32_16x16x32_bf16 ----------------
// Block = 256 threads = 8 wave32. Wave w computes rows [blk*128 + w*16, +16) x all 64 cols.
#define LDSTRIDE 65   // u32 stride per 128-elem bf16 row (64 pairs + 1 pad) -> breaks bank conflicts
__global__ __launch_bounds__(256) void gat_gemm(const float* __restrict__ x,
                                                const unsigned int* __restrict__ Wtu,
                                                float* __restrict__ h) {
  __shared__ unsigned int sW[HC * LDSTRIDE];         // 16,640 B
  __shared__ unsigned int sX[8][16 * LDSTRIDE];      // 33,280 B

  const int tid  = threadIdx.x;
  const int lane = tid & 31;
  const int wave = tid >> 5;
  const int nodeBase = blockIdx.x * 128;

  // Stage Wt (bf16 pairs) cooperatively: 64 rows x 64 pairs
  for (int i = tid; i < HC * 64; i += 256) {
    int n = i >> 6, kp = i & 63;
    sW[n * LDSTRIDE + kp] = Wtu[i];
  }
  // Stage this wave's 16 x-rows as bf16 pairs; one full row (512B) per iteration = coalesced
  {
    for (int r = 0; r < 16; ++r) {
      int node = nodeBase + wave * 16 + r;
      float4 f = make_float4(0.f, 0.f, 0.f, 0.f);
      if (node < N_NODES)
        f = *(const float4*)(x + (size_t)node * IN_CH + lane * 4);
      unsigned int* dst = &sX[wave][r * LDSTRIDE + lane * 2];
      dst[0] = pack_bf2(f.x, f.y);
      dst[1] = pack_bf2(f.z, f.w);
    }
  }
  __syncthreads();

  // CDNA5 16-bit A/B fragment layout (ISA 7.12.2): lanes 0-15 take K {0..7,16..23},
  // lanes 16-31 take K {8..15,24..31}; VGPR p holds K pair pmap[p] (in pair units).
  const int m    = lane & 15;
  const int kofs = (lane >> 4) * 4;            // pair offset: +4 pairs (=8 elems) for hi half
  const int pmap[8] = {0, 1, 2, 3, 8, 9, 10, 11};

  AccFrag acc[4];
#pragma unroll
  for (int nt = 0; nt < 4; ++nt) acc[nt].v = (v8f){0.f,0.f,0.f,0.f,0.f,0.f,0.f,0.f};

  const unsigned int* sXw = &sX[wave][0];
#pragma unroll
  for (int kt = 0; kt < 4; ++kt) {             // K = 128 in 4 steps of 32
    const int kpBase = kt * 16 + kofs;         // in bf16-pair units
    ABFrag a;
#pragma unroll
    for (int p = 0; p < 8; ++p)
      a.u[p] = sXw[m * LDSTRIDE + kpBase + pmap[p]];
#pragma unroll
    for (int nt = 0; nt < 4; ++nt) {
      ABFrag b;
      const int n = nt * 16 + m;
#pragma unroll
      for (int p = 0; p < 8; ++p)
        b.u[p] = sW[n * LDSTRIDE + kpBase + pmap[p]];
      acc[nt].v = __builtin_amdgcn_wmma_f32_16x16x32_bf16(
          false, a.v, false, b.v, (short)0, acc[nt].v, false, false);
    }
  }

  // C/D layout: lane holds col n = nt*16 + m; VGPR r holds row r + 8*(lane>=16)
  const int mhi = (lane >> 4) * 8;
#pragma unroll
  for (int nt = 0; nt < 4; ++nt) {
#pragma unroll
    for (int r = 0; r < 8; ++r) {
      int node = nodeBase + wave * 16 + r + mhi;
      if (node < N_NODES)
        h[(size_t)node * HC + nt * 16 + m] = acc[nt].f[r];
    }
  }
}

// ---------------- K3: per-(node,head) alpha_l/alpha_r dots + init amax/denom ----------------
__global__ void gat_node_prep(const float* __restrict__ h, const float* __restrict__ a,
                              float* __restrict__ al, float* __restrict__ ar,
                              float* __restrict__ amax, float* __restrict__ denom) {
  int i = blockIdx.x * blockDim.x + threadIdx.x;     // i = node*4 + head
  if (i >= N_NODES * HEADS) return;
  int n = i >> 2, hd = i & 3;
  const float* hp  = h + (size_t)n * HC + hd * OUT_CH;
  const float* avL = a + (size_t)hd * 2 * OUT_CH;
  const float* avR = avL + OUT_CH;
  float sl = 0.f, sr = 0.f;
#pragma unroll
  for (int c = 0; c < OUT_CH; ++c) { float v = hp[c]; sl += v * avL[c]; sr += v * avR[c]; }
  al[i] = sl; ar[i] = sr;
  amax[i]  = -INFINITY;     // re-initialized every call -> deterministic across graph replays
  denom[i] = 0.f;
}

// ---------------- K4: edge pass 1 — alpha + leaky relu + segment max ----------------
__global__ void gat_edge_pass1(const int* __restrict__ ei, const float* __restrict__ ea,
                               const float* __restrict__ Wedge,
                               const float* __restrict__ al, const float* __restrict__ ar,
                               float* __restrict__ alphaE, float* __restrict__ amax) {
  int e = blockIdx.x * blockDim.x + threadIdx.x;
  if (e >= N_EDGES) return;
  int row = ei[e];
  int col = ei[N_EDGES + e];
  const float* eap = ea + (size_t)e * EDGE_CH;
  float acc0 = 0.f, acc1 = 0.f, acc2 = 0.f, acc3 = 0.f;
#pragma unroll
  for (int j = 0; j < EDGE_CH; ++j) {
    float v = eap[j];
    const float* wr = Wedge + j * HEADS;
    acc0 += v * wr[0]; acc1 += v * wr[1]; acc2 += v * wr[2]; acc3 += v * wr[3];
  }
  float acc[HEADS] = {acc0, acc1, acc2, acc3};
#pragma unroll
  for (int hd = 0; hd < HEADS; ++hd) {
    float aval = al[row * HEADS + hd] + ar[col * HEADS + hd] + acc[hd];
    aval = aval > 0.f ? aval : 0.2f * aval;
    alphaE[(size_t)e * HEADS + hd] = aval;
    atomicMaxF(&amax[col * HEADS + hd], aval);
  }
}

// ---------------- K5: edge pass 2 — exp + segment sum ----------------
__global__ void gat_edge_pass2(const int* __restrict__ ei, const float* __restrict__ alphaE,
                               const float* __restrict__ amax, float* __restrict__ denom) {
  int e = blockIdx.x * blockDim.x + threadIdx.x;
  if (e >= N_EDGES) return;
  int col = ei[N_EDGES + e];
#pragma unroll
  for (int hd = 0; hd < HEADS; ++hd) {
    float ex = __expf(alphaE[(size_t)e * HEADS + hd] - amax[col * HEADS + hd]);
    atomicAdd(&denom[col * HEADS + hd], ex);
  }
}

// ---------------- K6: out[n] = h[n] * denom/clip(denom)  (exact segment-sum algebra) ----------------
__global__ void gat_output(const float* __restrict__ h, const float* __restrict__ denom,
                           float* __restrict__ out) {
  int i = blockIdx.x * blockDim.x + threadIdx.x;     // i = node*4 + head
  if (i >= N_NODES * HEADS) return;
  float d = denom[i];
  float s = d / fmaxf(d, 1e-10f);                    // d==0 (no in-edges) -> 0
  int n = i >> 2, hd = i & 3;
  const float4* hp = (const float4*)(h + (size_t)n * HC + hd * OUT_CH);
  float4* op = (float4*)(out + (size_t)n * HC + hd * OUT_CH);
#pragma unroll
  for (int c = 0; c < OUT_CH / 4; ++c) {
    float4 v = hp[c];
    v.x *= s; v.y *= s; v.z *= s; v.w *= s;
    op[c] = v;
  }
}

// ---------------- launcher ----------------
extern "C" void kernel_launch(void* const* d_in, const int* in_sizes, int n_in,
                              void* d_out, int out_size, void* d_ws, size_t ws_size,
                              hipStream_t stream) {
  const float* x     = (const float*)d_in[0];
  const int*   ei    = (const int*)d_in[1];
  const float* ea    = (const float*)d_in[2];
  const float* W     = (const float*)d_in[3];
  const float* Wedge = (const float*)d_in[4];
  const float* a     = (const float*)d_in[5];
  float* out = (float*)d_out;

  char* ws = (char*)d_ws;
  float*        h      = (float*)(ws + WS_H);
  unsigned int* Wtu    = (unsigned int*)(ws + WS_WT);
  float*        al     = (float*)(ws + WS_AL);
  float*        ar     = (float*)(ws + WS_AR);
  float*        amax   = (float*)(ws + WS_AMAX);
  float*        denom  = (float*)(ws + WS_DENOM);
  float*        alphaE = (float*)(ws + WS_ALPHAE);

  const int wtN   = HC * (IN_CH / 2);
  const int nhN   = N_NODES * HEADS;
  const int gemmB = (N_NODES + 127) / 128;

  gat_wt_convert<<<(wtN + 255) / 256, 256, 0, stream>>>(W, Wtu);
  gat_gemm<<<gemmB, 256, 0, stream>>>(x, Wtu, h);
  gat_node_prep<<<(nhN + 255) / 256, 256, 0, stream>>>(h, a, al, ar, amax, denom);
  gat_edge_pass1<<<(N_EDGES + 255) / 256, 256, 0, stream>>>(ei, ea, Wedge, al, ar, alphaE, amax);
  gat_edge_pass2<<<(N_EDGES + 255) / 256, 256, 0, stream>>>(ei, alphaE, amax, denom);
  gat_output<<<(nhN + 255) / 256, 256, 0, stream>>>(h, denom, out);
}
